// CompoundClassifier_18279380812530
// MI455X (gfx1250) — matrix operand order repaired and checked by reading, compiled, not run on hardware
//
#include <hip/hip_runtime.h>

#define HIDDEN   128
#define EDGES_PER_BLOCK 128
#define THREADS  128              // 4 waves; each wave owns two 16-edge M-tiles
#define LDS_CHUNKS 16             // 16 chunks of 8 halves (16B) per 128-half row

typedef _Float16 v16h __attribute__((ext_vector_type(16)));
typedef _Float16 v8h  __attribute__((ext_vector_type(8)));
typedef float    v8f  __attribute__((ext_vector_type(8)));

// -------- prep A: fp32 feature tables -> f16 --------
__global__ void cvt_f16_kernel(const float* __restrict__ xi, int nIngElems,
                               const float* __restrict__ xc, int nCmpElems,
                               _Float16* __restrict__ s16, _Float16* __restrict__ d16) {
    int tid = blockIdx.x * blockDim.x + threadIdx.x;
    if (tid < nIngElems) {
        s16[tid] = (_Float16)xi[tid];
    } else if (tid < nIngElems + nCmpElems) {
        int t = tid - nIngElems;
        d16[t] = (_Float16)xc[t];
    }
}

// -------- prep B: pack W1[256][128] row-major into WMMA B-fragment order --------
// frag element j of lane L for (nt,kc):  N = nt*16 + (L%16),
//   K = kc*32 + (L<16 ? 0 : 16) + j      (contiguous k order, same as A)
__global__ void pack_w1_kernel(const float* __restrict__ W1, _Float16* __restrict__ w1pack) {
    int tid = blockIdx.x * blockDim.x + threadIdx.x;   // 8*8*32*16 = 32768 threads
    if (tid >= 8 * 8 * 32 * 16) return;
    int j    = tid & 15;
    int lane = (tid >> 4) & 31;
    int kc   = (tid >> 9) & 7;
    int nt   = (tid >> 12) & 7;
    int n = nt * 16 + (lane & 15);
    int k = kc * 32 + ((lane < 16) ? 0 : 16) + j;
    w1pack[tid] = (_Float16)W1[k * HIDDEN + n];
}

// XOR-swizzled LDS address (half-element index). Rows are 128 halves = 16 chunks
// of 16B; chunk index is XORed with (row & 15) so 16 consecutive rows at the same
// logical chunk hit 16 distinct bank groups.
__device__ __forceinline__ int swz(int row, int chunk) {
    return row * HIDDEN + ((chunk ^ (row & 15)) << 3);
}

// -------- main: 128 edges per block, 4 waves, each wave = two 16-edge M-tiles ----
__global__ __launch_bounds__(THREADS, 1)
void edge_mlp_kernel(const _Float16* __restrict__ srcF16,
                     const _Float16* __restrict__ dstF16,
                     const _Float16* __restrict__ w1pack,
                     const int* __restrict__ edge_src,
                     const int* __restrict__ edge_dst,
                     const float* __restrict__ b1,
                     const float* __restrict__ W2,
                     const float* __restrict__ b2,
                     float* __restrict__ out, int E) {
    __shared__ _Float16 Asrc[EDGES_PER_BLOCK * HIDDEN];   // 32 KB
    __shared__ _Float16 Adst[EDGES_PER_BLOCK * HIDDEN];   // 32 KB

    const int tid    = threadIdx.x;
    const int blockE = blockIdx.x * EDGES_PER_BLOCK;

    // ---- gather: each thread loads one src row and one dst row ----
    {
        int r  = tid;                       // 0..127
        int e  = blockE + r;
        int eC = (e < E) ? e : (E - 1);
        int is = edge_src[eC];
        int id = edge_dst[eC];
        const float4* gs = (const float4*)(srcF16 + (size_t)is * HIDDEN);
        const float4* gd = (const float4*)(dstF16 + (size_t)id * HIDDEN);
        #pragma unroll
        for (int c = 0; c < LDS_CHUNKS; ++c) {
            float4 vs = gs[c];
            float4 vd = gd[c];
            *(float4*)(Asrc + swz(r, c)) = vs;
            *(float4*)(Adst + swz(r, c)) = vd;
        }
    }
    __syncthreads();

    const int wave  = tid >> 5;                    // 0..3 -> M-tile pair
    const int lane  = tid & 31;
    const int mrow0 = wave * 32 + (lane & 15);     // A row, first M-tile
    const int mrow1 = mrow0 + 16;                  // A row, second M-tile
    const int abase = (lane < 16) ? 0 : 16;        // K sub-offset (contiguous k order, same as B)

    v8f acc0[8] = {};                              // M-tile 0: 8 N-tiles of 16x16 f32
    v8f acc1[8] = {};                              // M-tile 1

    const v16h* wp = (const v16h*)w1pack;

    union AF { v16h v; v8h h[2]; };
    AF a0, a1, a0n, a1n;

    // A layout: lane holds K = base+{0..15} (elems 0-15)
    {
        int c0 = abase >> 3;                       // kc = 0
        a0.h[0] = *(const v8h*)(Asrc + swz(mrow0, c0));
        a0.h[1] = *(const v8h*)(Asrc + swz(mrow0, c0 + 1));
        a1.h[0] = *(const v8h*)(Asrc + swz(mrow1, c0));
        a1.h[1] = *(const v8h*)(Asrc + swz(mrow1, c0 + 1));
    }

    #pragma unroll
    for (int kc = 0; kc < 8; ++kc) {
        // batch-preload the 8 B fragments for this K-chunk
        v16h bfr[8];
        #pragma unroll
        for (int nt = 0; nt < 8; ++nt)
            bfr[nt] = wp[(nt * 8 + kc) * 32 + lane];

        // prefetch next A fragments (LDS) behind the WMMA chain
        if (kc < 7) {
            int kcn = kc + 1;
            const _Float16* Abuf = (kcn < 4) ? Asrc : Adst;
            int c0 = (((kcn & 3) * 32) + abase) >> 3;
            a0n.h[0] = *(const v8h*)(Abuf + swz(mrow0, c0));
            a0n.h[1] = *(const v8h*)(Abuf + swz(mrow0, c0 + 1));
            a1n.h[0] = *(const v8h*)(Abuf + swz(mrow1, c0));
            a1n.h[1] = *(const v8h*)(Abuf + swz(mrow1, c0 + 1));
        }

        // 16 WMMAs; each B fragment feeds two consecutive WMMAs
        #pragma unroll
        for (int nt = 0; nt < 8; ++nt) {
            acc0[nt] = __builtin_amdgcn_wmma_f32_16x16x32_f16(
                false, a0.v, false, bfr[nt], (short)0, acc0[nt], false, false);
            acc1[nt] = __builtin_amdgcn_wmma_f32_16x16x32_f16(
                false, a1.v, false, bfr[nt], (short)0, acc1[nt], false, false);
        }
        a0 = a0n;
        a1 = a1n;
    }

    // ---- epilogue: +b1, ReLU, dot with W2 per row, reduce across 16-lane N group ----
    const int ncol = lane & 15;
    float p0[8], p1[8];
    #pragma unroll
    for (int r = 0; r < 8; ++r) { p0[r] = 0.0f; p1[r] = 0.0f; }

    #pragma unroll
    for (int nt = 0; nt < 8; ++nt) {
        int n    = nt * 16 + ncol;
        float w2 = W2[n];
        float bb = b1[n];
        #pragma unroll
        for (int r = 0; r < 8; ++r) {
            float h0 = acc0[nt][r] + bb;
            float h1 = acc1[nt][r] + bb;
            h0 = h0 > 0.0f ? h0 : 0.0f;
            h1 = h1 > 0.0f ? h1 : 0.0f;
            p0[r] += h0 * w2;
            p1[r] += h1 * w2;
        }
    }
    #pragma unroll
    for (int off = 1; off < 16; off <<= 1) {
        #pragma unroll
        for (int r = 0; r < 8; ++r) {
            p0[r] += __shfl_xor(p0[r], off, 32);
            p1[r] += __shfl_xor(p1[r], off, 32);
        }
    }

    if ((lane & 15) == 0) {
        // D layout: lanes 0-15 elem r -> row r; lanes 16-31 elem r -> row r+8
        int mbase0 = blockE + wave * 32 + ((lane >= 16) ? 8 : 0);
        int mbase1 = mbase0 + 16;
        float bias2 = b2[0];
        #pragma unroll
        for (int r = 0; r < 8; ++r) {
            int e0 = mbase0 + r;
            int e1 = mbase1 + r;
            if (e0 < E) {
                float x = p0[r] + bias2;
                out[e0] = 1.0f / (1.0f + __expf(-x));
            }
            if (e1 < E) {
                float x = p1[r] + bias2;
                out[e1] = 1.0f / (1.0f + __expf(-x));
            }
        }
    }
}

extern "C" void kernel_launch(void* const* d_in, const int* in_sizes, int n_in,
                              void* d_out, int out_size, void* d_ws, size_t ws_size,
                              hipStream_t stream) {
    const float* x_ing = (const float*)d_in[0];
    const float* x_cmp = (const float*)d_in[1];
    const int*   edges = (const int*)d_in[2];
    const float* W1    = (const float*)d_in[3];
    const float* b1    = (const float*)d_in[4];
    const float* W2    = (const float*)d_in[5];
    const float* b2    = (const float*)d_in[6];
    float* out = (float*)d_out;

    const int nIngElems = in_sizes[0];           // N_ING * 128
    const int nCmpElems = in_sizes[1];           // N_CMP * 128
    const int E         = in_sizes[2] / 2;
    const int* edge_src = edges;
    const int* edge_dst = edges + E;

    // workspace layout (256B aligned chunks)
    char* ws = (char*)d_ws;
    size_t o = 0;
    _Float16* srcF16 = (_Float16*)(ws + o); o += ((size_t)nIngElems * 2 + 255) & ~(size_t)255;
    _Float16* dstF16 = (_Float16*)(ws + o); o += ((size_t)nCmpElems * 2 + 255) & ~(size_t)255;
    _Float16* w1pack = (_Float16*)(ws + o); o += (8 * 8 * 32 * 16 * 2);
    (void)ws_size; (void)n_in; (void)out_size;

    // prep
    {
        int total = nIngElems + nCmpElems;
        int blocks = (total + 255) / 256;
        cvt_f16_kernel<<<blocks, 256, 0, stream>>>(x_ing, nIngElems, x_cmp, nCmpElems,
                                                   srcF16, dstF16);
        pack_w1_kernel<<<(8 * 8 * 32 * 16) / 256, 256, 0, stream>>>(W1, w1pack);
    }

    // main
    {
        int blocks = (E + EDGES_PER_BLOCK - 1) / EDGES_PER_BLOCK;
        edge_mlp_kernel<<<blocks, THREADS, 0, stream>>>(srcF16, dstF16, w1pack,
                                                        edge_src, edge_dst,
                                                        b1, W2, b2, out, E);
    }
}
